// OnlineItemSimilarity_16801912062162
// MI455X (gfx1250) — matrix operand order repaired and checked by reading, not compiled, not run on hardware
//
#include <hip/hip_runtime.h>
#include <hip/hip_bf16.h>

typedef __attribute__((ext_vector_type(16))) _Float16 v16h;
typedef __attribute__((ext_vector_type(8)))  _Float16 v8h;
typedef __attribute__((ext_vector_type(4)))  _Float16 v4h;
typedef __attribute__((ext_vector_type(8)))  float    v8f;

#define N_ITEMS   500000
#define EMBED_DIM 64
#define BATCH     1024
#define N_TILES   (N_ITEMS / 16)        // 31250 item tiles of 16
#define N_CHUNKS  16
#define TPC       ((N_TILES + N_CHUNKS - 1) / N_CHUNKS)   // 1954 tiles per chunk
#define Q_TILES   (BATCH / 16)          // 64 query tiles

#define NEG_INF   (-__builtin_inff())

// ---------------------------------------------------------------------------
// Kernel 1: convert the full embedding table fp32 -> fp16 (64 MB, fits in L2).
// ---------------------------------------------------------------------------
__global__ __launch_bounds__(256)
void convert_f32_to_f16_kernel(const float* __restrict__ emb,
                               _Float16* __restrict__ ehalf) {
    size_t i      = (size_t)blockIdx.x * blockDim.x + threadIdx.x;
    size_t stride = (size_t)gridDim.x * blockDim.x;
    const size_t n4 = (size_t)N_ITEMS * EMBED_DIM / 4;   // 8M float4 groups
    for (; i < n4; i += stride) {
        float4 f = ((const float4*)emb)[i];
        v4h h = { (_Float16)f.x, (_Float16)f.y, (_Float16)f.z, (_Float16)f.w };
        ((v4h*)ehalf)[i] = h;
    }
}

// ---------------------------------------------------------------------------
// Kernel 2: gather the 1024 query rows (idx = item_idx - 1, with JAX-style
// negative wraparound) and pack them into the per-lane B-operand fragment
// layout of v_wmma_f32_16x16x32_f16:
//   frag f (K = 32f..32f+31): lanes 0-15 -> N = lane,    K = 32f+0..7, 32f+16..23
//                             lanes 16-31 -> N = lane-16, K = 32f+8..15, 32f+24..31
// Stored so lane `l` of query-tile `t`, frag `f` reads 32 contiguous bytes.
// ---------------------------------------------------------------------------
__global__ __launch_bounds__(256)
void gather_pack_queries_kernel(const float* __restrict__ emb,
                                const int* __restrict__ item_idx,
                                _Float16* __restrict__ qfrag) {
    int tid  = blockIdx.x * blockDim.x + threadIdx.x;   // 0..2047
    if (tid >= Q_TILES * 32) return;
    int t    = tid >> 5;
    int lane = tid & 31;
    int n    = t * 16 + (lane & 15);
    long long self = (long long)item_idx[n] - 1;
    if (self < 0) self += N_ITEMS;                      // JAX negative index wrap
    const float* row = emb + self * (long long)EMBED_DIM;
    int khalf = (lane < 16) ? 0 : 8;
#pragma unroll
    for (int f = 0; f < 2; ++f) {
        int ks = f * 32 + khalf;
        float4 r0 = *(const float4*)(row + ks);
        float4 r1 = *(const float4*)(row + ks + 4);
        float4 r2 = *(const float4*)(row + ks + 16);
        float4 r3 = *(const float4*)(row + ks + 20);
        v16h h = { (_Float16)r0.x, (_Float16)r0.y, (_Float16)r0.z, (_Float16)r0.w,
                   (_Float16)r1.x, (_Float16)r1.y, (_Float16)r1.z, (_Float16)r1.w,
                   (_Float16)r2.x, (_Float16)r2.y, (_Float16)r2.z, (_Float16)r2.w,
                   (_Float16)r3.x, (_Float16)r3.y, (_Float16)r3.z, (_Float16)r3.w };
        *(v16h*)(qfrag + ((size_t)(2 * t + f) * 32 + lane) * 16) = h;
    }
}

// ---------------------------------------------------------------------------
// Helpers for kernel 3.
// ---------------------------------------------------------------------------
__device__ __forceinline__ void load_tile_frags(const _Float16* __restrict__ ehalf,
                                                int t, int l16, int k0,
                                                v16h& a0, v16h& a1) {
    // A-operand lanes read two contiguous 16-byte chunks of a row-major row:
    //   lane<16:  M=lane,    chunks at K=0..7  and K=16..23 (+32 for frag 1)
    //   lane>=16: M=lane-16, chunks at K=8..15 and K=24..31 (+32 for frag 1)
    const _Float16* row = ehalf + ((size_t)t * 16 + l16) * EMBED_DIM;
    union U { v16h v; v8h h[2]; };
    U u0, u1;
    u0.h[0] = *(const v8h*)(row + k0);
    u0.h[1] = *(const v8h*)(row + k0 + 16);
    u1.h[0] = *(const v8h*)(row + k0 + 32);
    u1.h[1] = *(const v8h*)(row + k0 + 48);
    a0 = u0.v;
    a1 = u1.v;
}

__device__ __forceinline__ void epilogue(v8f c, int itemBase, int selfIdx,
                                         float& best, int& bestIdx) {
    // Rare branch 1: self item falls in this lane's 8-row strip on <=1 tile of
    // the whole scan; common path is sub + cmp + execz-skipped branch.
    unsigned du = (unsigned)(selfIdx - itemBase);
    if (du < 8u) {
#pragma unroll
        for (int r = 0; r < 8; ++r)
            if (du == (unsigned)r) c[r] = NEG_INF;
    }
    // Value-only max tree (fuses to v_max3_num_f32), ~4 VALU.
    float m01 = fmaxf(c[0], c[1]);
    float m23 = fmaxf(c[2], c[3]);
    float m45 = fmaxf(c[4], c[5]);
    float m67 = fmaxf(c[6], c[7]);
    float tile = fmaxf(fmaxf(m01, m23), fmaxf(m45, m67));
    // Rare branch 2: recover the argmax index only when the tile wins (decays
    // like 1/t over the scan).
    if (tile > best) {
        best = tile;
        int bi = itemBase;
#pragma unroll
        for (int r = 1; r < 8; ++r)
            if (c[r] == tile) bi = itemBase + r;
        bestIdx = bi;
    }
}

#define WMMA_F16(A, B, C) \
    __builtin_amdgcn_wmma_f32_16x16x32_f16(false, (A), false, (B), (short)0, (C), false, false)

// ---------------------------------------------------------------------------
// Kernel 3: fused GEMM + running max/argmax.
//   D(16 items x 16 queries) = A(items, K=64) x B(K=64, queries).
//   Two independent item tiles per iteration -> 4 interleaved WMMAs with no
//   D->C RAW between adjacent WMMAs (kills the hazard v_nops).
//   D layout: lane covers one query column (N = lane&15) and 8 item rows
//   (M = 0..7 for lane<16, 8..15 otherwise) -> per-lane scalar max/argmax.
// ---------------------------------------------------------------------------
__global__ __launch_bounds__(256)
void sim_argmax_kernel(const _Float16* __restrict__ ehalf,
                       const _Float16* __restrict__ qfrag,
                       const int* __restrict__ item_idx,
                       float* __restrict__ pval,
                       int* __restrict__ pidx) {
    const int qtile = blockIdx.x;            // 0..63
    const int chunk = blockIdx.y;            // 0..15
    const int tid   = threadIdx.x;
    const int lane  = tid & 31;
    const int wave  = __builtin_amdgcn_readfirstlane(tid >> 5);  // scalar wave id
    const int l16   = lane & 15;

    const int q = qtile * 16 + l16;
    long long self = (long long)item_idx[q] - 1;
    if (self < 0) self += N_ITEMS;
    const int selfIdx = (int)self;

    // Query (B-operand) fragments: resident in VGPRs for the whole loop.
    v16h b0 = *(const v16h*)(qfrag + ((size_t)(2 * qtile + 0) * 32 + lane) * 16);
    v16h b1 = *(const v16h*)(qfrag + ((size_t)(2 * qtile + 1) * 32 + lane) * 16);

    const int k0    = (lane < 16) ? 0 : 8;   // A-fragment chunk-0 K base
    const int mbase = (lane < 16) ? 0 : 8;   // D rows covered by this lane

    float best    = NEG_INF;
    int   bestIdx = 0;

    const int tBeg = chunk * TPC;
    const int tLim = (tBeg + TPC < N_TILES) ? (tBeg + TPC) : N_TILES;

    int t = tBeg + wave;                     // stride-8 slot domain, paired 2x
    for (; t + 8 < tLim; t += 16) {          // wave-uniform bounds: EXEC all-1s
        v16h a0t, a1t, a0u, a1u;
        load_tile_frags(ehalf, t,     l16, k0, a0t, a1t);
        load_tile_frags(ehalf, t + 8, l16, k0, a0u, a1u);
        __builtin_prefetch(ehalf + ((size_t)(t + 16) * 16 + l16) * EMBED_DIM, 0, 3);

        v8f c0 = {}, c1 = {};
        c0 = WMMA_F16(a0t, b0, c0);
        c1 = WMMA_F16(a0u, b0, c1);          // independent: fills hazard slots
        c0 = WMMA_F16(a1t, b1, c0);
        c1 = WMMA_F16(a1u, b1, c1);

        epilogue(c0, t * 16 + mbase,       selfIdx, best, bestIdx);
        epilogue(c1, (t + 8) * 16 + mbase, selfIdx, best, bestIdx);
    }
    if (t < tLim) {                          // tail: single tile
        v16h a0t, a1t;
        load_tile_frags(ehalf, t, l16, k0, a0t, a1t);
        v8f c0 = {};
        c0 = WMMA_F16(a0t, b0, c0);
        c0 = WMMA_F16(a1t, b1, c0);
        epilogue(c0, t * 16 + mbase, selfIdx, best, bestIdx);
    }

    // Block reduction: combine 8 waves and the (lane, lane+16) column pair.
    __shared__ float sBest[256];
    __shared__ int   sIdx[256];
    sBest[tid] = best;
    sIdx[tid]  = bestIdx;
    __syncthreads();
    if (tid < 16) {
        float b = NEG_INF; int bi = 0;
#pragma unroll
        for (int w = 0; w < 8; ++w) {
#pragma unroll
            for (int h = 0; h < 2; ++h) {
                int s = w * 32 + tid + 16 * h;
                if (sBest[s] > b) { b = sBest[s]; bi = sIdx[s]; }
            }
        }
        int qq = qtile * 16 + tid;
        pval[qq * N_CHUNKS + chunk] = b;
        pidx[qq * N_CHUNKS + chunk] = bi;
    }
}

// ---------------------------------------------------------------------------
// Kernel 4: final reduction over chunks + monotonic normalization; write
// d_out = [indices(1024) | scores(1024)] as floats.
// ---------------------------------------------------------------------------
__global__ __launch_bounds__(256)
void final_reduce_kernel(const float* __restrict__ pval,
                         const int* __restrict__ pidx,
                         const float* __restrict__ min_score,
                         const float* __restrict__ max_score,
                         float* __restrict__ out) {
    int q = blockIdx.x * blockDim.x + threadIdx.x;
    if (q >= BATCH) return;
    float best = NEG_INF; int bi = 0;
#pragma unroll
    for (int c = 0; c < N_CHUNKS; ++c) {
        float v = pval[q * N_CHUNKS + c];
        if (v > best) { best = v; bi = pidx[q * N_CHUNKS + c]; }
    }
    float mn = min_score[0], mx = max_score[0];
    out[q]         = (float)(bi + 1);                  // argmax + 1
    out[BATCH + q] = (best - mn) / (mx - mn);          // normalized score
}

// ---------------------------------------------------------------------------
extern "C" void kernel_launch(void* const* d_in, const int* in_sizes, int n_in,
                              void* d_out, int out_size, void* d_ws, size_t ws_size,
                              hipStream_t stream) {
    const float* emb       = (const float*)d_in[0];    // (500000, 64) f32
    const int*   item_idx  = (const int*)d_in[1];      // (1024,) int
    const float* min_score = (const float*)d_in[2];    // (1,)
    const float* max_score = (const float*)d_in[3];    // (1,)
    float*       out       = (float*)d_out;            // 2048 floats

    // Workspace layout
    char* ws = (char*)d_ws;
    const size_t EHALF_BYTES = (size_t)N_ITEMS * EMBED_DIM * sizeof(_Float16);   // 64 MB
    const size_t QFRAG_BYTES = (size_t)Q_TILES * 2 * 32 * 16 * sizeof(_Float16); // 128 KB
    const size_t PVAL_BYTES  = (size_t)BATCH * N_CHUNKS * sizeof(float);
    _Float16* ehalf = (_Float16*)ws;
    _Float16* qfrag = (_Float16*)(ws + EHALF_BYTES);
    float*    pval  = (float*)(ws + EHALF_BYTES + QFRAG_BYTES);
    int*      pidx  = (int*)(ws + EHALF_BYTES + QFRAG_BYTES + PVAL_BYTES);

    convert_f32_to_f16_kernel<<<2048, 256, 0, stream>>>(emb, ehalf);
    gather_pack_queries_kernel<<<(Q_TILES * 32 + 255) / 256, 256, 0, stream>>>(
        emb, item_idx, qfrag);
    sim_argmax_kernel<<<dim3(Q_TILES, N_CHUNKS), 256, 0, stream>>>(
        ehalf, qfrag, item_idx, pval, pidx);
    final_reduce_kernel<<<(BATCH + 255) / 256, 256, 0, stream>>>(
        pval, pidx, min_score, max_score, out);
}
